// SVDAttention_86612310491926
// MI455X (gfx1250) — matrix operand order repaired and checked by reading, not compiled, not run on hardware
//
#include <hip/hip_runtime.h>
#include <hip/hip_bf16.h>

typedef __attribute__((ext_vector_type(16))) __bf16 v16bf;
typedef __attribute__((ext_vector_type(8)))  float  v8f;

#define B_  4
#define H_  8
#define N_  4096
#define R_  64
#define BH_ 32

union ABFrag { v16bf v; unsigned int u[8]; };

__device__ __forceinline__ unsigned short f2bf(float f) {
    unsigned int x = __float_as_uint(f);
    x += 0x7FFFu + ((x >> 16) & 1u);        // round-to-nearest-even
    return (unsigned short)(x >> 16);
}

// A fragment: 16x32 bf16, element (m,k) at base[m*stride + k] (stride in ushorts)
__device__ __forceinline__ v16bf load_A(const unsigned short* base, int stride) {
    int lane = threadIdx.x & 31;
    int half = lane >> 4, m = lane & 15;
    ABFrag f;
#pragma unroll
    for (int i = 0; i < 8; ++i) {
        int k = ((i >> 2) << 4) + (half << 3) + ((i & 3) << 1);
        f.u[i] = *(const unsigned int*)(base + m * stride + k);
    }
    return f.v;
}

// B fragment: 32x16 bf16, element (k,n) at base[k*stride + n]
__device__ __forceinline__ v16bf load_B(const unsigned short* base, int stride) {
    int lane = threadIdx.x & 31;
    int half = lane >> 4, n = lane & 15;
    ABFrag f;
#pragma unroll
    for (int j = 0; j < 8; ++j) {
        int k = ((j >> 2) << 4) + (half << 3) + ((j & 3) << 1);
        f.u[j] = (unsigned int)base[k * stride + n] | ((unsigned int)base[(k + 1) * stride + n] << 16);
    }
    return f.v;
}

__device__ __forceinline__ void store_C(float* dst, int stride, const v8f& c) {
    int lane = threadIdx.x & 31;
    int half = lane >> 4, n = lane & 15;
#pragma unroll
    for (int g = 0; g < 8; ++g) dst[(g + 8 * half) * stride + n] = c[g];
}

__device__ __forceinline__ void atomic_C(float* dst, int stride, const v8f& c) {
    int lane = threadIdx.x & 31;
    int half = lane >> 4, n = lane & 15;
#pragma unroll
    for (int g = 0; g < 8; ++g) atomicAdd(&dst[(g + 8 * half) * stride + n], c[g]);
}

__device__ __forceinline__ v8f wmma_bf16(v16bf a, v16bf b, v8f c) {
    return __builtin_amdgcn_wmma_f32_16x16x32_bf16(false, a, false, b, (short)0, c,
                                                   false, false);
}

// ---------------- kernel 1: zero accumulators ----------------
__global__ void kzero(float* p, int n) {
    int i = blockIdx.x * blockDim.x + threadIdx.x;
    if (i < n) p[i] = 0.0f;
}

// ---------------- kernel 2: per-chunk online column (max, sumexp) ----------------
__global__ __launch_bounds__(256) void kstats(const float* __restrict__ svdV,
                                              const float* __restrict__ mask,
                                              float* __restrict__ pmax,
                                              float* __restrict__ psum) {
    int bh = blockIdx.y, chunk = blockIdx.x;    // 8 chunks of 512 rows
    int b = bh >> 3;
    int t = threadIdx.x, col = t & 63, grp = t >> 6;
    const float* base = svdV + (size_t)bh * N_ * R_;
    float m = -__builtin_inff(), s = 0.0f;
    int row0 = chunk * 512 + grp * 128;
    for (int i = 0; i < 128; ++i) {
        int n = row0 + i;
        float x = base[(size_t)n * R_ + col] - 1e9f * (1.0f - mask[b * N_ + n]);
        float nm = fmaxf(m, x);
        s = s * __expf(m - nm) + __expf(x - nm);
        m = nm;
    }
    __shared__ float sm[4][64], ss[4][64];
    sm[grp][col] = m; ss[grp][col] = s;
    __syncthreads();
    if (grp == 0) {
        for (int g2 = 1; g2 < 4; ++g2) {
            float m2 = sm[g2][col], s2 = ss[g2][col];
            float nm = fmaxf(m, m2);
            s = s * __expf(m - nm) + s2 * __expf(m2 - nm);
            m = nm;
        }
        pmax[(bh * 8 + chunk) * 64 + col] = m;
        psum[(bh * 8 + chunk) * 64 + col] = s;
    }
}

// ---------------- kernel 3: merge chunk partials ----------------
__global__ void kmerge(const float* __restrict__ pmax, const float* __restrict__ psum,
                       float* __restrict__ cmax, float* __restrict__ cinv) {
    int bh = blockIdx.x, r = threadIdx.x;
    float m = -__builtin_inff(), s = 0.0f;
    for (int c = 0; c < 8; ++c) {
        float m2 = pmax[(bh * 8 + c) * 64 + r], s2 = psum[(bh * 8 + c) * 64 + r];
        float nm = fmaxf(m, m2);
        s = s * __expf(m - nm) + s2 * __expf(m2 - nm);
        m = nm;
    }
    cmax[bh * 64 + r] = m;
    cinv[bh * 64 + r] = 1.0f / s;
}

// ---------------- kernel 4: graph filter = poly(sigmoid(Sigma)) ----------------
__global__ void kfilt(const float* __restrict__ Sigma, const float* __restrict__ gammas,
                      float* __restrict__ filt) {
    int bh = blockIdx.x, r = threadIdx.x;
    float s = 1.0f / (1.0f + __expf(-Sigma[bh * 64 + r]));
    float g = gammas[4];
    for (int k = 3; k >= 0; --k) g = g * s + gammas[k];
    filt[bh * 64 + r] = g;
}

// ---------------- kernel 5: KtV = K^T V and GramK = K^T K via WMMA ----------------
// 8 waves: waves 0-3 -> KtV row-blocks 0-3, waves 4-7 -> GramK row-blocks 0-3.
// Each wave: one shared A fragment per step, 4 B fragments, 4 in-place WMMAs.
__global__ __launch_bounds__(256) void kaccum(const float* __restrict__ svdV,
                                              const float* __restrict__ Vmat,
                                              const float* __restrict__ mask,
                                              const float* __restrict__ cmax,
                                              const float* __restrict__ cinv,
                                              float* __restrict__ KtV,
                                              float* __restrict__ GramK) {
    int bh = blockIdx.y, chunk = blockIdx.x;
    int b = bh >> 3;
    int t = threadIdx.x, w = t >> 5;
    __shared__ unsigned short Kn[32][64];   // n-major K tile (B operand)
    __shared__ unsigned short Kr[64][32];   // r-major K tile (A operand)
    __shared__ unsigned short Vs[32][64];   // V tile (B operand)
    __shared__ float scm[64], sci[64];
    if (t < 64) { scm[t] = cmax[bh * 64 + t]; sci[t] = cinv[bh * 64 + t]; }
    __syncthreads();
    const float* svb = svdV + (size_t)bh * N_ * R_;
    const float* vb  = Vmat + (size_t)bh * N_ * R_;

    int ti = w & 3;
    bool isGram = (w >= 4);
    const unsigned short* Bbase = isGram ? &Kn[0][0] : &Vs[0][0];   // uniform select
    const unsigned short* Abase = &Kr[ti * 16][0];

    v8f acc0, acc1, acc2, acc3;
#pragma unroll
    for (int e = 0; e < 8; ++e) { acc0[e] = 0.f; acc1[e] = 0.f; acc2[e] = 0.f; acc3[e] = 0.f; }

    for (int step = 0; step < 16; ++step) {
        int n0 = chunk * 512 + step * 32;
#pragma unroll
        for (int i = 0; i < 8; ++i) {
            int idx = t + i * 256;
            int row = idx >> 6, col = idx & 63;
            int n = n0 + row;
            float x = svb[(size_t)n * R_ + col] - 1e9f * (1.0f - mask[b * N_ + n]);
            unsigned short kb = f2bf(__expf(x - scm[col]) * sci[col]);
            Kn[row][col] = kb;
            Kr[col][row] = kb;
            Vs[row][col] = f2bf(vb[(size_t)n * R_ + col]);
        }
        __syncthreads();
        v16bf a = load_A(Abase, 32);
        acc0 = wmma_bf16(a, load_B(Bbase + 0,  64), acc0);
        acc1 = wmma_bf16(a, load_B(Bbase + 16, 64), acc1);
        acc2 = wmma_bf16(a, load_B(Bbase + 32, 64), acc2);
        acc3 = wmma_bf16(a, load_B(Bbase + 48, 64), acc3);
        __syncthreads();
    }
    float* dst = (isGram ? GramK : KtV) + (size_t)bh * 4096 + (ti * 16) * 64;
    atomic_C(dst + 0,  64, acc0);
    atomic_C(dst + 16, 64, acc1);
    atomic_C(dst + 32, 64, acc2);
    atomic_C(dst + 48, 64, acc3);
}

// ---------------- kernel 6: X = (Q*filt) @ KtV, GramQ = Q^T Q ----------------
__global__ __launch_bounds__(256) void kxout(const float* __restrict__ U,
                                             const float* __restrict__ KtV,
                                             const float* __restrict__ filt,
                                             float* __restrict__ Xout,
                                             float* __restrict__ GramQ) {
    int bh = blockIdx.y, tile = blockIdx.x;  // 32 tiles of 128 rows
    int t = threadIdx.x, w = t >> 5;
    __shared__ unsigned short Bs[64][64];    // filt-scaled KtV, bf16
    __shared__ unsigned short Qn[32][64];    // n-major Q tile
    __shared__ unsigned short Qt[64][32];    // r-major Q tile
#pragma unroll
    for (int i = 0; i < 16; ++i) {
        int idx = t + i * 256;
        int rr = idx >> 6;
        Bs[0][idx] = f2bf(filt[bh * 64 + rr] * KtV[(size_t)bh * 4096 + idx]);
    }
    v8f gacc0, gacc1;
#pragma unroll
    for (int e = 0; e < 8; ++e) { gacc0[e] = 0.f; gacc1[e] = 0.f; }

    const float* ub = U + (size_t)bh * N_ * R_;
    int row = t >> 3, l8 = t & 7;            // 8 lanes per row (aligned in wave32)

    // per-wave tile assignments (uniform)
    int ib = w >> 2, jj = w & 3;             // X tile
    int gti = w >> 1, gtj0 = (w & 1) * 2;    // GramQ tiles gti x {gtj0, gtj0+1}

    __syncthreads();                         // Bs ready

    for (int step = 0; step < 4; ++step) {
        int n0 = tile * 128 + step * 32;
        float v[8];
        float m = -__builtin_inff();
#pragma unroll
        for (int i = 0; i < 8; ++i) {
            v[i] = ub[(size_t)(n0 + row) * R_ + l8 + 8 * i];
            m = fmaxf(m, v[i]);
        }
#pragma unroll
        for (int o = 1; o < 8; o <<= 1) m = fmaxf(m, __shfl_xor(m, o, 8));
        float s = 0.0f;
#pragma unroll
        for (int i = 0; i < 8; ++i) { v[i] = __expf(v[i] - m); s += v[i]; }
#pragma unroll
        for (int o = 1; o < 8; o <<= 1) s += __shfl_xor(s, o, 8);
        float inv = 1.0f / s;
#pragma unroll
        for (int i = 0; i < 8; ++i) {
            unsigned short qb = f2bf(v[i] * inv);
            Qn[row][l8 + 8 * i] = qb;
            Qt[l8 + 8 * i][row] = qb;
        }
        __syncthreads();

        // X tile (K-dim = 64, two WMMAs, in-place accumulate from zero)
        v8f xc;
#pragma unroll
        for (int e = 0; e < 8; ++e) xc[e] = 0.0f;
        xc = wmma_bf16(load_A(&Qn[ib * 16][0],  64), load_B(&Bs[0][jj * 16],  64), xc);
        xc = wmma_bf16(load_A(&Qn[ib * 16][32], 64), load_B(&Bs[32][jj * 16], 64), xc);
        float* xdst = Xout + (size_t)bh * N_ * R_ + (size_t)(n0 + ib * 16) * 64 + jj * 16;
        store_C(xdst, 64, xc);

        // GramQ accumulation: shared A fragment, two B fragments
        v16bf ga = load_A(&Qt[gti * 16][0], 32);
        gacc0 = wmma_bf16(ga, load_B(&Qn[0][gtj0 * 16],       64), gacc0);
        gacc1 = wmma_bf16(ga, load_B(&Qn[0][(gtj0 + 1) * 16], 64), gacc1);
        __syncthreads();
    }
    float* gdst = GramQ + (size_t)bh * 4096 + (gti * 16) * 64;
    atomic_C(gdst + gtj0 * 16,       64, gacc0);
    atomic_C(gdst + (gtj0 + 1) * 16, 64, gacc1);
}

// ---------------- kernel 7: ortho loss ----------------
__global__ __launch_bounds__(256) void kloss(const float* __restrict__ GramQ,
                                             const float* __restrict__ GramK,
                                             float* __restrict__ out) {
    int b = blockIdx.x, t = threadIdx.x;
    float sum = 0.0f;
    for (int idx = t; idx < 8 * 4096; idx += 256) {
        int h = idx >> 12, e = idx & 4095;
        int r = e >> 6, s = e & 63;
        float eye = (r == s) ? 1.0f : 0.0f;
        size_t o = (size_t)(b * 8 + h) * 4096 + e;
        sum += fabsf(GramQ[o] - eye) + fabsf(GramK[o] - eye);
    }
    __shared__ float red[256];
    red[t] = sum;
    __syncthreads();
    for (int o = 128; o > 0; o >>= 1) {
        if (t < o) red[t] += red[t + o];
        __syncthreads();
    }
    if (t == 0) out[b] = 0.1f * red[0] / (8.0f * 64.0f * 64.0f);
}

extern "C" void kernel_launch(void* const* d_in, const int* in_sizes, int n_in,
                              void* d_out, int out_size, void* d_ws, size_t ws_size,
                              hipStream_t stream) {
    const float* U     = (const float*)d_in[0];
    const float* Sigma = (const float*)d_in[1];
    const float* svdV  = (const float*)d_in[2];
    const float* Vmat  = (const float*)d_in[3];
    const float* mask  = (const float*)d_in[4];
    const float* gam   = (const float*)d_in[5];

    float* ws    = (float*)d_ws;
    float* KtV   = ws;                 // 32*4096
    float* GramK = ws + 131072;        // 32*4096
    float* GramQ = ws + 262144;        // 32*4096
    float* pmax  = ws + 393216;        // 32*8*64
    float* psum  = ws + 409600;        // 32*8*64
    float* cmax  = ws + 425984;        // 32*64
    float* cinv  = ws + 428032;        // 32*64
    float* filt  = ws + 430080;        // 32*64

    float* Xout = (float*)d_out;
    float* loss = (float*)d_out + (size_t)B_ * H_ * N_ * R_;

    kzero<<<(393216 + 255) / 256, 256, 0, stream>>>(ws, 393216);
    kstats<<<dim3(8, BH_), 256, 0, stream>>>(svdV, mask, pmax, psum);
    kmerge<<<BH_, 64, 0, stream>>>(pmax, psum, cmax, cinv);
    kfilt<<<BH_, 64, 0, stream>>>(Sigma, gam, filt);
    kaccum<<<dim3(8, BH_), 256, 0, stream>>>(svdV, Vmat, mask, cmax, cinv, KtV, GramK);
    kxout<<<dim3(32, BH_), 256, 0, stream>>>(U, KtV, filt, Xout, GramQ);
    kloss<<<B_, 256, 0, stream>>>(GramQ, GramK, loss);
}
